// TopkMoeFFN_42434276884752
// MI455X (gfx1250) — matrix operand order repaired and checked by reading, not compiled, not run on hardware
//
#include <hip/hip_runtime.h>
#include <hip/hip_bf16.h>
#include <math.h>

// ---------------------------------------------------------------------------
// Top-2 MoE FFN, fused: gating (WMMA bf16) + dense-expert WMMA + combine.
//   N=131072 tokens, H=O=128, E=8 experts, top_k=2.
// One wave (32 lanes, wave32) owns a 16-token tile. All expert weights are
// staged once per workgroup into LDS as bf16, transposed so B fragments are
// contiguous per lane.  Matmuls: v_wmma_f32_16x16x32_bf16, f32 accumulate.
// Round 7: __builtin_amdgcn_sched_barrier(0) between the batched B-fragment
// ds_load_b128s and the WMMA K-chain, so the scheduler cannot sink the loads
// onto their consumers and the waits become graduated (pipelined) instead of
// four serial s_wait_dscnt 0x0 stalls.
// ---------------------------------------------------------------------------

typedef __bf16 bf16x16 __attribute__((ext_vector_type(16)));
typedef float  f32x8   __attribute__((ext_vector_type(8)));
typedef float  f32x4   __attribute__((ext_vector_type(4)));
typedef unsigned int   u32x8  __attribute__((ext_vector_type(8)));
typedef unsigned int   u32x4  __attribute__((ext_vector_type(4)));

#define HDIM 128
#define ODIM 128
#define NEXP 8
#define KP   136                       // padded row pitch in bf16 elems (16B-aligned, de-conflicted)
#define WT_ELEMS (NEXP * ODIM * KP)    // transposed expert weights  (bf16)
#define WG_ELEMS (16 * KP)             // transposed padded gate weights (bf16)
#define SMEM_BYTES (WT_ELEMS * 2 + WG_ELEMS * 2 + NEXP * ODIM * 4 + 8 * 16 * NEXP * 4)

static __device__ __forceinline__ void sched_fence() {
#if __has_builtin(__builtin_amdgcn_sched_barrier)
  __builtin_amdgcn_sched_barrier(0);
#endif
}

static __device__ __forceinline__ unsigned pack2bf(float a, float b) {
  // dst[15:0] = bf16(a), dst[31:16] = bf16(b), round-to-nearest-even
  unsigned r;
  asm("v_cvt_pk_bf16_f32 %0, %1, %2" : "=v"(r) : "v"(a), "v"(b));
  return r;
}

static __device__ __forceinline__ bf16x16 load_bfrag(const unsigned short* p) {
  // 32 contiguous bytes from an LDS weight row -> 16x(K=32) B fragment (per-lane K range)
  u32x4 a = *(const u32x4*)(p);
  u32x4 b = *(const u32x4*)(p + 8);
  u32x8 v;
  v[0] = a[0]; v[1] = a[1]; v[2] = a[2]; v[3] = a[3];
  v[4] = b[0]; v[5] = b[1]; v[6] = b[2]; v[7] = b[3];
  return __builtin_bit_cast(bf16x16, v);
}

__global__ void __launch_bounds__(256)
__attribute__((amdgpu_waves_per_eu(2)))
moe_topk_ffn_kernel(const float* __restrict__ x,
                    const float* __restrict__ w_gate,
                    const float* __restrict__ w_expert,
                    const float* __restrict__ b_expert,
                    float* __restrict__ out,
                    int nTokens)
{
  extern __shared__ char smem[];
  unsigned short* wt  = (unsigned short*)smem;          // [E][O][KP]  W_e^T, bf16
  unsigned short* wg  = wt + WT_ELEMS;                  // [16][KP]    w_gate^T padded, bf16
  float*          bl  = (float*)(wg + WG_ELEMS);        // [E][O]      bias
  float*          gsc = bl + NEXP * ODIM;               // [8 waves][16][E] logits->gates

  const int tid = threadIdx.x;

  // ---- stage expert weights: transpose + f32->bf16 (coalesced global reads) ----
  for (int idx = tid; idx < NEXP * 64 * ODIM; idx += 256) {
    int n = idx & 127;
    int t = idx >> 7;
    int k = (t & 63) << 1;
    int e = t >> 6;
    float f0 = w_expert[(e * HDIM + k) * ODIM + n];
    float f1 = w_expert[(e * HDIM + k + 1) * ODIM + n];
    *(unsigned*)&wt[(e * ODIM + n) * KP + k] = pack2bf(f0, f1);
  }
  // ---- stage gate weights: transpose, pad cols 8..15 with zeros ----
  for (int idx = tid; idx < 16 * 64; idx += 256) {
    int n = idx & 15;
    int k = (idx >> 4) << 1;
    float f0 = (n < NEXP) ? w_gate[k * NEXP + n] : 0.f;
    float f1 = (n < NEXP) ? w_gate[(k + 1) * NEXP + n] : 0.f;
    *(unsigned*)&wg[n * KP + k] = pack2bf(f0, f1);
  }
  // ---- stage bias ----
  for (int idx = tid; idx < NEXP * ODIM; idx += 256) bl[idx] = b_expert[idx];
  __syncthreads();

  const int lane = tid & 31;
  const int wv   = tid >> 5;
  const int ln16 = lane & 15;
  const int hh   = lane >> 4;          // lane half selects K sub-range
  float* g = gsc + wv * (16 * NEXP);

  const int wavesTotal = gridDim.x * 8;
  const int waveId     = blockIdx.x * 8 + wv;
  const int nTiles     = nTokens >> 4;

  for (int tile = waveId; tile < nTiles; tile += wavesTotal) {
    const int row = tile * 16 + ln16;
    const float* xr = x + (long)row * HDIM;

    // ---- A fragments: 4 K-steps of 16x32 bf16 (ISA A layout: lane=row) ----
    bf16x16 afrag[4];
#pragma unroll
    for (int ks = 0; ks < 4; ++ks) {
      int kb = ks * 32 + hh * 16;
      f32x4 q0 = __builtin_nontemporal_load((const f32x4*)(xr + kb));
      f32x4 q1 = __builtin_nontemporal_load((const f32x4*)(xr + kb + 4));
      f32x4 q2 = __builtin_nontemporal_load((const f32x4*)(xr + kb + 8));
      f32x4 q3 = __builtin_nontemporal_load((const f32x4*)(xr + kb + 12));
      u32x8 v;
      v[0] = pack2bf(q0[0], q0[1]); v[1] = pack2bf(q0[2], q0[3]);
      v[2] = pack2bf(q1[0], q1[1]); v[3] = pack2bf(q1[2], q1[3]);
      v[4] = pack2bf(q2[0], q2[1]); v[5] = pack2bf(q2[2], q2[3]);
      v[6] = pack2bf(q3[0], q3[1]); v[7] = pack2bf(q3[2], q3[3]);
      afrag[ks] = __builtin_bit_cast(bf16x16, v);
    }

    // ---- prefetch next tile's x slab (8 KB) while this tile computes ----
    if (tile + wavesTotal < nTiles) {
      const char* pnext = (const char*)(x + (long)(tile + wavesTotal) * 16 * HDIM);
      __builtin_prefetch(pnext + lane * 256, 0, 0);
      __builtin_prefetch(pnext + lane * 256 + 128, 0, 0);
    }

    // ---- gating: logits(16x16, cols>=8 are zero) = A @ w_gate ----
    f32x8 lc;
#pragma unroll
    for (int i = 0; i < 8; ++i) lc[i] = 0.f;
    {
      bf16x16 bg[4];
#pragma unroll
      for (int ks = 0; ks < 4; ++ks)
        bg[ks] = load_bfrag(wg + ln16 * KP + ks * 32 + hh * 16);
      sched_fence();   // keep all 8 ds_load_b128 issued before the WMMA chain
#pragma unroll
      for (int ks = 0; ks < 4; ++ks)
        lc = __builtin_amdgcn_wmma_f32_16x16x32_bf16(false, afrag[ks], false, bg[ks],
                                                     (short)0, lc, false, false);
    }

    // scatter logits (C layout) to per-wave LDS scratch
    if (ln16 < NEXP) {
#pragma unroll
      for (int i = 0; i < 8; ++i) g[(i + hh * 8) * NEXP + ln16] = lc[i];
    }
    asm volatile("s_wait_dscnt 0x0" ::: "memory");

    // ---- per-token top-2 + softmax over top-2 (lane t < 16 handles token t) ----
    if (lane < 16) {
      float v[NEXP];
#pragma unroll
      for (int j = 0; j < NEXP; ++j) v[j] = g[lane * NEXP + j];
      int i1 = 0;
#pragma unroll
      for (int j = 1; j < NEXP; ++j) if (v[j] > v[i1]) i1 = j;
      int i2 = (i1 == 0) ? 1 : 0;
#pragma unroll
      for (int j = 0; j < NEXP; ++j) if (j != i1 && v[j] > v[i2]) i2 = j;
      float ex  = __expf(v[i2] - v[i1]);     // <= 1
      float g2v = ex / (1.f + ex);
      float g1v = 1.f - g2v;
#pragma unroll
      for (int j = 0; j < NEXP; ++j)
        g[lane * NEXP + j] = (j == i1) ? g1v : ((j == i2) ? g2v : 0.f);
    }
    asm volatile("s_wait_dscnt 0x0" ::: "memory");

    // ---- dense expert sweep, gate-weighted combine (gates mostly zero) ----
    float oacc[64];
#pragma unroll
    for (int i = 0; i < 64; ++i) oacc[i] = 0.f;

    for (int e = 0; e < NEXP; ++e) {
      float gv[8];
#pragma unroll
      for (int i = 0; i < 8; ++i) gv[i] = g[(i + hh * 8) * NEXP + e];
      const unsigned short* wte = wt + e * (ODIM * KP);
#pragma unroll
      for (int nt = 0; nt < 8; ++nt) {
        int n = nt * 16 + ln16;
        float bb = bl[e * ODIM + n];
        bf16x16 bf[4];
#pragma unroll
        for (int ks = 0; ks < 4; ++ks)
          bf[ks] = load_bfrag(wte + n * KP + ks * 32 + hh * 16);
        sched_fence(); // keep all 8 ds_load_b128 issued before the WMMA chain
        f32x8 acc;
#pragma unroll
        for (int i = 0; i < 8; ++i) acc[i] = 0.f;
#pragma unroll
        for (int ks = 0; ks < 4; ++ks)
          acc = __builtin_amdgcn_wmma_f32_16x16x32_bf16(false, afrag[ks], false, bf[ks],
                                                        (short)0, acc, false, false);
#pragma unroll
        for (int i = 0; i < 8; ++i) oacc[nt * 8 + i] += gv[i] * (acc[i] + bb);
      }
    }

    // ---- store 16x128 f32 output tile (streaming, non-temporal) ----
#pragma unroll
    for (int nt = 0; nt < 8; ++nt) {
#pragma unroll
      for (int i = 0; i < 8; ++i) {
        __builtin_nontemporal_store(
            oacc[nt * 8 + i],
            out + (long)(tile * 16 + i + hh * 8) * ODIM + nt * 16 + ln16);
      }
    }
  }
}

extern "C" void kernel_launch(void* const* d_in, const int* in_sizes, int n_in,
                              void* d_out, int out_size, void* d_ws, size_t ws_size,
                              hipStream_t stream) {
  (void)n_in; (void)out_size; (void)d_ws; (void)ws_size;
  const float* x   = (const float*)d_in[0];
  const float* wgt = (const float*)d_in[1];
  // d_in[2] = w_noise: inactive in eval-mode gating
  const float* wex = (const float*)d_in[3];
  const float* bex = (const float*)d_in[4];
  // d_in[5] = top_k (== 2, hardwired in the top-2 path)
  float* out = (float*)d_out;

  int nTokens = in_sizes[0] / HDIM;

  (void)hipFuncSetAttribute((const void*)moe_topk_ffn_kernel,
                            hipFuncAttributeMaxDynamicSharedMemorySize, SMEM_BYTES);
  moe_topk_ffn_kernel<<<256, 256, SMEM_BYTES, stream>>>(x, wgt, wex, bex, out, nTokens);
}
